// MoEProcessor_33595234189785
// MI455X (gfx1250) — matrix operand order repaired and checked by reading, not compiled, not run on hardware
//
#include <hip/hip_runtime.h>
#include <math.h>

// ---------------- problem constants (from reference) ----------------
#define BATCH 4
#define SEQ   2048
#define TOK   (BATCH * SEQ)   // 8192 tokens
#define DIM   1024
#define NEXP  8
#define LN_EPS 1e-5f
#define CAP   TOK             // per-expert slot capacity (max possible)
// GEMM tiling: block tile 128 tokens x 128 features, 8 waves (4M x 2N),
// each wave computes a 32x64 C slab = 8 16x16 accumulators.
#define TILE_M 128
#define TILE_N 128
#define MT2   (CAP / TILE_M)  // 64 worst-case M tiles per expert
#define NT2   (DIM / TILE_N)  // 8 N tiles

// ---------------- CDNA5 WMMA vector types ----------------
typedef __attribute__((ext_vector_type(16))) __bf16 v16bf;
typedef __attribute__((ext_vector_type(8)))  __bf16 v8bf;
typedef __attribute__((ext_vector_type(8)))  float  v8f;

union AFrag { v16bf v; v8bf h[2]; };

__device__ __forceinline__ unsigned short f2bf(float f) {
    unsigned int u = __float_as_uint(f);
    u += 0x7FFFu + ((u >> 16) & 1u);   // round-to-nearest-even
    return (unsigned short)(u >> 16);
}

// ---------------- kernel 1: x fp32 -> bf16 (row major) ----------------
__global__ __launch_bounds__(256) void cvt_x_kernel(const float* __restrict__ x,
                                                    unsigned short* __restrict__ xb) {
    size_t i = ((size_t)blockIdx.x * 256 + threadIdx.x) * 4;
    float4 v = *(const float4*)(x + i);
    unsigned long long p = (unsigned long long)f2bf(v.x)
                         | ((unsigned long long)f2bf(v.y) << 16)
                         | ((unsigned long long)f2bf(v.z) << 32)
                         | ((unsigned long long)f2bf(v.w) << 48);
    *(unsigned long long*)(xb + i) = p;
}

// ------- kernel 2: W_exp[e][d][f] fp32 -> Wt[e][f][d] bf16 (transpose) -------
__global__ __launch_bounds__(256) void cvt_w_kernel(const float* __restrict__ W,
                                                    unsigned short* __restrict__ Wt) {
    __shared__ float tile[32][33];
    int e  = blockIdx.z;
    int d0 = blockIdx.y * 32;
    int f0 = blockIdx.x * 32;
    const float* src = W + ((size_t)e << 20);
    for (int i = threadIdx.y; i < 32; i += 8)
        tile[i][threadIdx.x] = src[(size_t)(d0 + i) * DIM + f0 + threadIdx.x];
    __syncthreads();
    unsigned short* dst = Wt + ((size_t)e << 20);
    for (int i = threadIdx.y; i < 32; i += 8)
        dst[(size_t)(f0 + i) * DIM + d0 + threadIdx.x] = f2bf(tile[threadIdx.x][i]);
}

// ---------------- kernel 3: routing (one wave32 per token) ----------------
__global__ __launch_bounds__(256) void routing_kernel(
    const float* __restrict__ x, const float* __restrict__ noise,
    const float* __restrict__ Wr, const float* __restrict__ br,
    const float* __restrict__ gamma, const float* __restrict__ beta,
    int* __restrict__ counts, int* __restrict__ slotTok, float* __restrict__ slotW) {
    int wave = threadIdx.x >> 5;
    int lane = threadIdx.x & 31;
    int t = blockIdx.x * 8 + wave;
    const float* xr = x + (size_t)t * DIM;

    float acc[NEXP];
#pragma unroll
    for (int e = 0; e < NEXP; ++e) acc[e] = 0.0f;

    for (int i = 0; i < DIM / 32; ++i) {
        int d = lane + i * 32;
        float xv = xr[d];
        const float* wr = Wr + (size_t)d * NEXP;
#pragma unroll
        for (int e = 0; e < NEXP; ++e) acc[e] = fmaf(xv, wr[e], acc[e]);
    }
#pragma unroll
    for (int m = 16; m >= 1; m >>= 1)
#pragma unroll
        for (int e = 0; e < NEXP; ++e) acc[e] += __shfl_xor(acc[e], m, 32);

    float lg[NEXP];
#pragma unroll
    for (int e = 0; e < NEXP; ++e) lg[e] = acc[e] + br[e];

    float mu = 0.0f;
#pragma unroll
    for (int e = 0; e < NEXP; ++e) mu += lg[e];
    mu *= (1.0f / NEXP);
    float var = 0.0f;
#pragma unroll
    for (int e = 0; e < NEXP; ++e) { float d = lg[e] - mu; var = fmaf(d, d, var); }
    var *= (1.0f / NEXP);
    float inv = rsqrtf(var + LN_EPS);

    float ln[NEXP];
#pragma unroll
    for (int e = 0; e < NEXP; ++e) ln[e] = (lg[e] - mu) * inv * gamma[e] + beta[e];

    float mx = ln[0];
#pragma unroll
    for (int e = 1; e < NEXP; ++e) mx = fmaxf(mx, ln[e]);
    float p[NEXP], s = 0.0f;
#pragma unroll
    for (int e = 0; e < NEXP; ++e) { p[e] = expf(ln[e] - mx); s += p[e]; }
    float is = 1.0f / s;
    float r[NEXP];
#pragma unroll
    for (int e = 0; e < NEXP; ++e) r[e] = p[e] * is + noise[(size_t)t * NEXP + e];

    int i0 = 0; float v0 = r[0];
#pragma unroll
    for (int e = 1; e < NEXP; ++e) if (r[e] > v0) { v0 = r[e]; i0 = e; }
    int i1 = (i0 == 0) ? 1 : 0; float v1 = r[i1];
#pragma unroll
    for (int e = 0; e < NEXP; ++e) if (e != i0 && r[e] > v1) { v1 = r[e]; i1 = e; }

    float e1 = expf(v1 - v0);
    float sw = 1.0f + e1;
    float w0 = 1.0f / sw;
    float w1 = e1 / sw;

    if (lane == 0) {
        int p0 = atomicAdd(&counts[i0], 1);
        slotTok[i0 * CAP + p0] = t;
        slotW [i0 * CAP + p0] = w0;
        int p1 = atomicAdd(&counts[i1], 1);
        slotTok[i1 * CAP + p1] = t;
        slotW [i1 * CAP + p1] = w1;
    }
}

// ------- kernel 4: grouped expert GEMM via bf16 WMMA, scatter-add out -------
#define LOAD_FRAGS(A0, A1, B0, B1, B2, B3, OFF)                     \
    A0.h[0] = *(const v8bf*)(ar0 + (OFF));                          \
    A0.h[1] = *(const v8bf*)(ar0 + (OFF) + 8);                      \
    A1.h[0] = *(const v8bf*)(ar1 + (OFF));                          \
    A1.h[1] = *(const v8bf*)(ar1 + (OFF) + 8);                      \
    B0 = *(const v16bf*)(bp0 + (OFF));                              \
    B1 = *(const v16bf*)(bp1 + (OFF));                              \
    B2 = *(const v16bf*)(bp2 + (OFF));                              \
    B3 = *(const v16bf*)(bp3 + (OFF));

#define WMMA8(A0, A1, B0, B1, B2, B3)                                                            \
    acc00 = __builtin_amdgcn_wmma_f32_16x16x32_bf16(false, A0.v, false, B0, (short)0, acc00, false, false); \
    acc01 = __builtin_amdgcn_wmma_f32_16x16x32_bf16(false, A0.v, false, B1, (short)0, acc01, false, false); \
    acc02 = __builtin_amdgcn_wmma_f32_16x16x32_bf16(false, A0.v, false, B2, (short)0, acc02, false, false); \
    acc03 = __builtin_amdgcn_wmma_f32_16x16x32_bf16(false, A0.v, false, B3, (short)0, acc03, false, false); \
    acc10 = __builtin_amdgcn_wmma_f32_16x16x32_bf16(false, A1.v, false, B0, (short)0, acc10, false, false); \
    acc11 = __builtin_amdgcn_wmma_f32_16x16x32_bf16(false, A1.v, false, B1, (short)0, acc11, false, false); \
    acc12 = __builtin_amdgcn_wmma_f32_16x16x32_bf16(false, A1.v, false, B2, (short)0, acc12, false, false); \
    acc13 = __builtin_amdgcn_wmma_f32_16x16x32_bf16(false, A1.v, false, B3, (short)0, acc13, false, false);

__global__ __launch_bounds__(256) void moe_gemm_kernel(
    const unsigned short* __restrict__ xb,   // [TOK][DIM] bf16
    const unsigned short* __restrict__ wt,   // [E][f][d] bf16 (transposed)
    const float* __restrict__ b_exp,         // [E][DIM]
    const int* __restrict__ counts,
    const int* __restrict__ slotTok,
    const float* __restrict__ slotW,
    float* __restrict__ out) {
    int bx  = blockIdx.x;
    int e   = bx / (MT2 * NT2);
    int rem = bx % (MT2 * NT2);
    int mt  = rem / NT2;
    int nt  = rem % NT2;
    int cnt = counts[e];
    if (mt * TILE_M >= cnt) return;

    __shared__ int   ls_tok[TILE_M];
    __shared__ float ls_w[TILE_M];
    int tid = threadIdx.x;
    {
        int row = mt * TILE_M + tid;           // tid < 128 fills, 128..255 idle
        if (tid < TILE_M) {
            if (row < cnt) { ls_tok[tid] = slotTok[e * CAP + row]; ls_w[tid] = slotW[e * CAP + row]; }
            else           { ls_tok[tid] = 0;                      ls_w[tid] = 0.0f; }
        }
    }
    __syncthreads();

    int lane  = tid & 31;
    int wave  = tid >> 5;
    int mwave = wave >> 1;                 // 0..3 -> 32-row slab
    int nwave = wave & 1;                  // 0..1 -> 64-col slab
    int hi    = lane >> 4;                 // 0 = lanes 0-15, 1 = lanes 16-31
    int l15   = lane & 15;

    const __bf16* xbb = (const __bf16*)xb;
    const __bf16* wtb = (const __bf16*)wt;

    // A rows for this wave's two 16-row groups (gathered tokens).
    // A k order chosen here (16x32 bf16, same as B): lanes 0-15: K k..k+15; lanes 16-31: k+16..k+31.
    int m0 = mwave * 32 + l15;
    const __bf16* ar0 = xbb + (size_t)ls_tok[m0]      * DIM + hi * 16;
    const __bf16* ar1 = xbb + (size_t)ls_tok[m0 + 16] * DIM + hi * 16;

    // B columns (k order chosen here, same as A): lanes 0-15: K k..k+15; lanes 16-31: k+16..k+31.
    int ncol = nt * TILE_N + nwave * 64 + l15;
    const __bf16* bp0 = wtb + ((size_t)e * DIM + ncol +  0) * DIM + hi * 16;
    const __bf16* bp1 = wtb + ((size_t)e * DIM + ncol + 16) * DIM + hi * 16;
    const __bf16* bp2 = wtb + ((size_t)e * DIM + ncol + 32) * DIM + hi * 16;
    const __bf16* bp3 = wtb + ((size_t)e * DIM + ncol + 48) * DIM + hi * 16;

    v8f acc00 = {}, acc01 = {}, acc02 = {}, acc03 = {};
    v8f acc10 = {}, acc11 = {}, acc12 = {}, acc13 = {};

    // Ping-pong double buffer over K: loads for one buffer are issued before
    // the 8 WMMAs consuming the other, so vmem overlaps matrix math.
    AFrag aX0, aX1, aY0, aY1;
    v16bf bX0, bX1, bX2, bX3, bY0, bY1, bY2, bY3;

    LOAD_FRAGS(aX0, aX1, bX0, bX1, bX2, bX3, 0)
    for (int k = 0; k < DIM; k += 64) {
        LOAD_FRAGS(aY0, aY1, bY0, bY1, bY2, bY3, k + 32)
        WMMA8(aX0, aX1, bX0, bX1, bX2, bX3)
        if (k + 64 < DIM) { LOAD_FRAGS(aX0, aX1, bX0, bX1, bX2, bX3, k + 64) }
        WMMA8(aY0, aY1, bY0, bY1, bY2, bY3)
    }

    // bias per owned column
    const float* be = b_exp + (size_t)e * DIM + ncol;
    float bias0 = be[0], bias1 = be[16], bias2 = be[32], bias3 = be[48];

    // C layout: VGPR r -> (M = r + 8*hi, N = l15). Scale by routing weight,
    // scatter-add; each (token, n) receives exactly 2 adds (commutative ->
    // bitwise deterministic).
#pragma unroll
    for (int mi = 0; mi < 2; ++mi) {
#pragma unroll
        for (int r = 0; r < 8; ++r) {
            int ml = mwave * 32 + mi * 16 + r + hi * 8;
            int tokm = ls_tok[ml];
            float wm = ls_w[ml];
            float* orow = out + (size_t)tokm * DIM + ncol;
            if (mi == 0) {
                atomicAdd(orow +  0, (acc00[r] + bias0) * wm);
                atomicAdd(orow + 16, (acc01[r] + bias1) * wm);
                atomicAdd(orow + 32, (acc02[r] + bias2) * wm);
                atomicAdd(orow + 48, (acc03[r] + bias3) * wm);
            } else {
                atomicAdd(orow +  0, (acc10[r] + bias0) * wm);
                atomicAdd(orow + 16, (acc11[r] + bias1) * wm);
                atomicAdd(orow + 32, (acc12[r] + bias2) * wm);
                atomicAdd(orow + 48, (acc13[r] + bias3) * wm);
            }
        }
    }
}

// ---------------- launcher ----------------
extern "C" void kernel_launch(void* const* d_in, const int* in_sizes, int n_in,
                              void* d_out, int out_size, void* d_ws, size_t ws_size,
                              hipStream_t stream) {
    (void)in_sizes; (void)n_in; (void)out_size; (void)ws_size;
    const float* x     = (const float*)d_in[0];
    const float* noise = (const float*)d_in[1];
    const float* Wr    = (const float*)d_in[2];
    const float* br    = (const float*)d_in[3];
    const float* gamma = (const float*)d_in[4];
    const float* beta  = (const float*)d_in[5];
    const float* Wexp  = (const float*)d_in[6];
    const float* bexp  = (const float*)d_in[7];
    float* out = (float*)d_out;

    char* ws = (char*)d_ws;
    const size_t XB_BYTES = (size_t)TOK * DIM * 2;          // 16 MiB
    const size_t WT_BYTES = (size_t)NEXP * DIM * DIM * 2;   // 16 MiB
    unsigned short* xb   = (unsigned short*)(ws);
    unsigned short* wt   = (unsigned short*)(ws + XB_BYTES);
    int*   counts  = (int*)  (ws + XB_BYTES + WT_BYTES);
    int*   slotTok = (int*)  (ws + XB_BYTES + WT_BYTES + 256);
    float* slotW   = (float*)(ws + XB_BYTES + WT_BYTES + 256 + (size_t)NEXP * CAP * 4);

    hipMemsetAsync(out, 0, (size_t)TOK * DIM * sizeof(float), stream);
    hipMemsetAsync(counts, 0, NEXP * sizeof(int), stream);

    cvt_x_kernel<<<(TOK * DIM) / (256 * 4), 256, 0, stream>>>(x, xb);
    cvt_w_kernel<<<dim3(DIM / 32, DIM / 32, NEXP), dim3(32, 8), 0, stream>>>(Wexp, wt);
    routing_kernel<<<TOK / 8, 256, 0, stream>>>(x, noise, Wr, br, gamma, beta,
                                                counts, slotTok, slotW);
    moe_gemm_kernel<<<NEXP * MT2 * NT2, 256, 0, stream>>>(xb, wt, bexp,
                                                          counts, slotTok, slotW, out);
}
